// CrossAttentionS2T_7722351198256
// MI455X (gfx1250) — compile-verified, matrix operand order checked by reading, not run on hardware
//
#include <hip/hip_runtime.h>
#include <math.h>

#define DIM   768
#define NHEAD 12
#define HD    64
#define B_    8
#define SPEC  4
#define T_    8
#define AP    196
#define VP    196
#define QLEN  (VP * T_)    // 1568
#define KLEN  (AP * SPEC)  // 784
#define MQ    (B_ * QLEN)  // 12544
#define MS    (B_ * KLEN)  // 6272
#define SCALE 0.125f

typedef __attribute__((ext_vector_type(16))) __bf16 v16bf;
typedef __attribute__((ext_vector_type(8)))  __bf16 v8bf;
typedef __attribute__((ext_vector_type(4)))  __bf16 v4bf;
typedef __attribute__((ext_vector_type(8)))  float  v8f;

__device__ __forceinline__ v8f zero8() {
  v8f z;
#pragma unroll
  for (int i = 0; i < 8; ++i) z[i] = 0.0f;
  return z;
}

// Async global->LDS copy, 16 bytes per active lane. Tracked by ASYNCcnt.
// LDS destination address = low 32 bits of the generic pointer (LDS aperture
// keeps the workgroup-relative offset in addr[31:0]).
__device__ __forceinline__ void async_load_b128(void* lds_ptr, const void* gptr) {
  uint32_t lds_off = (uint32_t)(uintptr_t)lds_ptr;
  uint64_t gaddr   = (uint64_t)(uintptr_t)gptr;
  asm volatile("global_load_async_to_lds_b128 %0, %1, off"
               :: "v"(lds_off), "v"(gaddr)
               : "memory");
}

__device__ __forceinline__ void wait_async(int n) {
  if (n == 0)
    asm volatile("s_wait_asynccnt 0" ::: "memory");
  else
    asm volatile("s_wait_asynccnt 6" ::: "memory");
}

// Load a 16x32 bf16 tile (rows = matrix rows, cols = contraction K) from a
// row-major buffer into the CDNA5 WMMA A-operand layout (also used for the
// B operand via the "load B^T row-major" trick).
// Lane L: row = L%16, K pattern = base8 + {0..7} and base8+16+{0..7}, base8 = 8*(L/16).
__device__ __forceinline__ v16bf load_frag(const __bf16* base, int stride, int lane) {
  const __bf16* p = base + (lane & 15) * stride + ((lane >> 4) << 3);
  v8bf lo = *(const v8bf*)(p);
  v8bf hi = *(const v8bf*)(p + 16);
  v16bf f;
#pragma unroll
  for (int i = 0; i < 8; ++i) { f[i] = lo[i]; f[i + 8] = hi[i]; }
  return f;
}

__device__ __forceinline__ v8f wmma_bf16(v16bf a, v16bf b, v8f c) {
  return __builtin_amdgcn_wmma_f32_16x16x32_bf16(false, a, false, b, (short)0, c, false, false);
}

// ---------------------------------------------------------------------------
// Prep: q_in[b, v*T+t, :] = t_x[1+v, b*T+t, :] + vmae_space_pos[v] + vmae_temporal_pos[t]
// ---------------------------------------------------------------------------
__global__ __launch_bounds__(256)
void prep_qin_kernel(const float* __restrict__ t_x, const float* __restrict__ spos,
                     const float* __restrict__ tpos, __bf16* __restrict__ out) {
  int gid = blockIdx.x * blockDim.x + threadIdx.x;  // MQ * 192 threads
  int m = gid / 192;
  int c = (gid % 192) * 4;
  int b = m / QLEN, q = m % QLEN;
  int v = q / T_, t = q % T_;
  float4 x  = *(const float4*)(t_x + ((size_t)(1 + v) * (B_ * T_) + b * T_ + t) * DIM + c);
  float4 sp = *(const float4*)(spos + (size_t)v * DIM + c);
  float4 tp = *(const float4*)(tpos + (size_t)t * DIM + c);
  v4bf o;
  o[0] = (__bf16)(x.x + sp.x + tp.x);
  o[1] = (__bf16)(x.y + sp.y + tp.y);
  o[2] = (__bf16)(x.z + sp.z + tp.z);
  o[3] = (__bf16)(x.w + sp.w + tp.w);
  *(v4bf*)(out + (size_t)m * DIM + c) = o;
}

// s[b, a*SPEC+p, :] = s_x[a, b*SPEC+p, :] + clip_space_pos[a] + clip_temporal_pos[p]
__global__ __launch_bounds__(256)
void prep_s_kernel(const float* __restrict__ s_x, const float* __restrict__ spos,
                   const float* __restrict__ tpos, __bf16* __restrict__ out) {
  int gid = blockIdx.x * blockDim.x + threadIdx.x;  // MS * 192 threads
  int m = gid / 192;
  int c = (gid % 192) * 4;
  int b = m / KLEN, q = m % KLEN;
  int a = q / SPEC, p = q % SPEC;
  float4 x  = *(const float4*)(s_x + ((size_t)a * (B_ * SPEC) + b * SPEC + p) * DIM + c);
  float4 sp = *(const float4*)(spos + (size_t)a * DIM + c);
  float4 tp = *(const float4*)(tpos + (size_t)p * DIM + c);
  v4bf o;
  o[0] = (__bf16)(x.x + sp.x + tp.x);
  o[1] = (__bf16)(x.y + sp.y + tp.y);
  o[2] = (__bf16)(x.z + sp.z + tp.z);
  o[3] = (__bf16)(x.w + sp.w + tp.w);
  *(v4bf*)(out + (size_t)m * DIM + c) = o;
}

__global__ __launch_bounds__(256)
void cvt_bf16_kernel(const float* __restrict__ in, __bf16* __restrict__ out, int n4) {
  int i = blockIdx.x * blockDim.x + threadIdx.x;
  if (i < n4) {
    float4 x = *(const float4*)(in + (size_t)i * 4);
    v4bf o;
    o[0] = (__bf16)x.x; o[1] = (__bf16)x.y; o[2] = (__bf16)x.z; o[3] = (__bf16)x.w;
    *(v4bf*)(out + (size_t)i * 4) = o;
  }
}

__global__ __launch_bounds__(256)
void cls_copy_kernel(const float* __restrict__ t_x, float* __restrict__ out) {
  int i = blockIdx.x * blockDim.x + threadIdx.x;  // (B_*T_*DIM)/4 threads
  *(float4*)(out + (size_t)i * 4) = *(const float4*)(t_x + (size_t)i * 4);
}

// ---------------------------------------------------------------------------
// GEMM: C[M,N] = A[M,K=768] @ W[N,K]^T + bias, bf16 inputs, f32 accumulate.
// MODE 0: out bf16, *SCALE (Q)   MODE 1: out bf16 (KV)
// MODE 2: out f32, scattered into (1+VP, B*T, DIM) layout (final projection)
// Block: 128 threads = 4 waves. Tile: 64(M) x 128(N), K-step 32.
// Double-buffered LDS; tiles staged with GLOBAL_LOAD_ASYNC_TO_LDS_B128 so the
// next tile's HBM/L2 traffic overlaps the current tile's WMMAs.
// ---------------------------------------------------------------------------
template <int MODE>
__global__ __launch_bounds__(128)
void gemm_bf16_kernel(const __bf16* __restrict__ A, const __bf16* __restrict__ W,
                      const float* __restrict__ bias, void* __restrict__ outp,
                      int M, int N) {
  __shared__ __align__(16) __bf16 As[2][64 * 40];
  __shared__ __align__(16) __bf16 Ws[2][128 * 40];
  const int tid = threadIdx.x;
  const int lane = tid & 31, wave = tid >> 5;
  const int wm = (wave & 1) * 32, wn = (wave >> 1) * 64;
  const int m0 = blockIdx.x * 64, n0 = blockIdx.y * 128;

  v8f acc[2][4];
#pragma unroll
  for (int i = 0; i < 2; ++i)
#pragma unroll
    for (int j = 0; j < 4; ++j) acc[i][j] = zero8();

  // Issue one stage (64x32 A chunks + 128x32 W chunks) as async copies.
  // Per thread: 2 + 4 b128 chunks -> 6 async instructions per wave per stage.
  auto issue_stage = [&](int buf, int k0) {
#pragma unroll
    for (int i = 0; i < 2; ++i) {
      int c = tid * 2 + i;                    // 0..255
      int r = c >> 2, cc = (c & 3) << 3;
      async_load_b128(&As[buf][r * 40 + cc],
                      A + (size_t)(m0 + r) * DIM + k0 + cc);
    }
#pragma unroll
    for (int i = 0; i < 4; ++i) {
      int c = tid * 4 + i;                    // 0..511
      int r = c >> 2, cc = (c & 3) << 3;
      async_load_b128(&Ws[buf][r * 40 + cc],
                      W + (size_t)(n0 + r) * DIM + k0 + cc);
    }
  };

  issue_stage(0, 0);

  const int KSTEPS = DIM / 32;  // 24
  for (int it = 0; it < KSTEPS; ++it) {
    const int cur = it & 1;
    if (it + 1 < KSTEPS) {
      issue_stage(cur ^ 1, (it + 1) * 32);
      wait_async(6);   // 6 newest outstanding = next stage; current stage done
    } else {
      wait_async(0);
    }
    __syncthreads();   // current buffer visible to all waves

    v16bf af[2], bf_[4];
#pragma unroll
    for (int i = 0; i < 2; ++i) af[i] = load_frag(&As[cur][(wm + i * 16) * 40], 40, lane);
#pragma unroll
    for (int j = 0; j < 4; ++j) bf_[j] = load_frag(&Ws[cur][(wn + j * 16) * 40], 40, lane);
#pragma unroll
    for (int i = 0; i < 2; ++i)
#pragma unroll
      for (int j = 0; j < 4; ++j) acc[i][j] = wmma_bf16(af[i], bf_[j], acc[i][j]);
    __syncthreads();   // done reading buf[cur] before next iter overwrites it
  }

  // Epilogue. C layout: lane L, vgpr v -> m = v + 8*(L/16), n = L%16.
#pragma unroll
  for (int i = 0; i < 2; ++i)
#pragma unroll
    for (int j = 0; j < 4; ++j)
#pragma unroll
      for (int v = 0; v < 8; ++v) {
        int ml = wm + i * 16 + v + ((lane >> 4) << 3);
        int nl = wn + j * 16 + (lane & 15);
        int gm = m0 + ml, gn = n0 + nl;
        float val = acc[i][j][v] + bias[gn];
        if (MODE == 0) {
          ((__bf16*)outp)[(size_t)gm * N + gn] = (__bf16)(val * SCALE);
        } else if (MODE == 1) {
          ((__bf16*)outp)[(size_t)gm * N + gn] = (__bf16)val;
        } else {
          int b = gm / QLEN, q = gm % QLEN;
          int vv = q / T_, tt = q % T_;
          size_t idx = ((size_t)(1 + vv) * (B_ * T_) + b * T_ + tt) * DIM + gn;
          ((float*)outp)[idx] = val;
        }
      }
}

// ---------------------------------------------------------------------------
// Flash attention. Grid: (B*NHEAD, 49). Block: 64 threads = 2 waves.
// Each wave owns one 16-row Q tile (98 tiles of 16 rows per (b,h)).
// K length 784 = 12 chunks of 64 + one chunk of 16 (NTC template = 4 or 1).
// K tile staged with async global->LDS; V^T transposed through LDS manually.
// ---------------------------------------------------------------------------
template <int NTC>
__device__ __forceinline__ void attn_chunk(
    int kbase, int b, int h, int tid, int lane, int wave,
    const __bf16* __restrict__ KVb, __bf16* Ks, __bf16* Vts, __bf16* Pw,
    const v16bf& qa0, const v16bf& qa1,
    v8f (&o)[4], float (&rmax)[8], float (&rsum)[8]) {
  const int valid = NTC * 16;
  // Cooperative stage: K tile [64 keys x 64 hd] and V^T tile [64 hd x 64 keys].
  {
    const int r = tid;  // 0..63
    if (r < valid) {
      const __bf16* krow = KVb + (size_t)(b * KLEN + kbase + r) * (2 * DIM) + h * HD;
      const __bf16* vrow = krow + DIM;
#pragma unroll
      for (int c2 = 0; c2 < 8; ++c2)
        async_load_b128(Ks + r * 72 + c2 * 8, krow + c2 * 8);
#pragma unroll
      for (int d = 0; d < 64; ++d) Vts[d * 72 + r] = vrow[d];
    } else {
      uint4 z = {0u, 0u, 0u, 0u};
#pragma unroll
      for (int c2 = 0; c2 < 8; ++c2) *(uint4*)(Ks + r * 72 + c2 * 8) = z;
#pragma unroll
      for (int d = 0; d < 64; ++d) Vts[d * 72 + r] = (__bf16)0.0f;
    }
  }
  wait_async(0);
  __syncthreads();

  // S = Q K^T  (contraction HD=64 -> 2 WMMA k-steps; NTC key tiles of 16)
  v8f s[NTC];
#pragma unroll
  for (int nt = 0; nt < NTC; ++nt) {
    s[nt] = zero8();
    v16bf kb0 = load_frag(Ks + nt * 16 * 72, 72, lane);
    s[nt] = wmma_bf16(qa0, kb0, s[nt]);
    v16bf kb1 = load_frag(Ks + nt * 16 * 72 + 32, 72, lane);
    s[nt] = wmma_bf16(qa1, kb1, s[nt]);
  }

  // Online softmax over this chunk, per row (8 rows/lane in C layout).
#pragma unroll
  for (int v = 0; v < 8; ++v) {
    float cm = -__builtin_inff();
#pragma unroll
    for (int nt = 0; nt < NTC; ++nt) cm = fmaxf(cm, s[nt][v]);
#pragma unroll
    for (int off = 8; off >= 1; off >>= 1) cm = fmaxf(cm, __shfl_xor(cm, off, 32));
    float nm = fmaxf(rmax[v], cm);
    float corr = __expf(rmax[v] - nm);
    rmax[v] = nm;
    rsum[v] *= corr;
#pragma unroll
    for (int j = 0; j < 4; ++j) o[j][v] *= corr;
    float psum = 0.0f;
#pragma unroll
    for (int nt = 0; nt < NTC; ++nt) {
      float p = __expf(s[nt][v] - nm);
      s[nt][v] = p;
      psum += p;
    }
#pragma unroll
    for (int off = 8; off >= 1; off >>= 1) psum += __shfl_xor(psum, off, 32);
    rsum[v] += psum;
    // Write P (C layout -> LDS row-major) for the A-operand transpose.
    int mrow = v + ((lane >> 4) << 3);
    int ncol = lane & 15;
#pragma unroll
    for (int nt = 0; nt < 4; ++nt)
      Pw[mrow * 72 + nt * 16 + ncol] = (nt < NTC) ? (__bf16)s[nt][v] : (__bf16)0.0f;
  }
  asm volatile("s_wait_dscnt 0" ::: "memory");  // P store -> P frag load (same wave)

  // O += P V  (contraction keys=64 -> 2 k-steps; 4 hd tiles of 16)
  v16bf pa0 = load_frag(Pw, 72, lane);
  v16bf pa1 = load_frag(Pw + 32, 72, lane);
#pragma unroll
  for (int j = 0; j < 4; ++j) {
    v16bf vb0 = load_frag(Vts + j * 16 * 72, 72, lane);
    o[j] = wmma_bf16(pa0, vb0, o[j]);
    v16bf vb1 = load_frag(Vts + j * 16 * 72 + 32, 72, lane);
    o[j] = wmma_bf16(pa1, vb1, o[j]);
  }
  __syncthreads();
}

__global__ __launch_bounds__(64)
void attention_kernel(const __bf16* __restrict__ Qb, const __bf16* __restrict__ KVb,
                      __bf16* __restrict__ Ob) {
  __shared__ __align__(16) __bf16 Ks[64 * 72];
  __shared__ __align__(16) __bf16 Vts[64 * 72];
  __shared__ __align__(16) __bf16 Ps[2][16 * 72];
  const int tid = threadIdx.x, lane = tid & 31, wave = tid >> 5;
  const int bh = blockIdx.x, b = bh / NHEAD, h = bh % NHEAD;
  const int qrow0 = (blockIdx.y * 2 + wave) * 16;

  const __bf16* qptr = Qb + (size_t)(b * QLEN + qrow0) * DIM + h * HD;
  v16bf qa0 = load_frag(qptr, DIM, lane);        // Q already has SCALE folded in
  v16bf qa1 = load_frag(qptr + 32, DIM, lane);

  v8f o[4];
#pragma unroll
  for (int j = 0; j < 4; ++j) o[j] = zero8();
  float rmax[8], rsum[8];
#pragma unroll
  for (int v = 0; v < 8; ++v) { rmax[v] = -__builtin_inff(); rsum[v] = 0.0f; }

  for (int ic = 0; ic < 12; ++ic)
    attn_chunk<4>(ic * 64, b, h, tid, lane, wave, KVb, Ks, Vts, &Ps[wave][0],
                  qa0, qa1, o, rmax, rsum);
  attn_chunk<1>(12 * 64, b, h, tid, lane, wave, KVb, Ks, Vts, &Ps[wave][0],
                qa0, qa1, o, rmax, rsum);

  // Normalize and store O (bf16, [MQ, DIM]) for the projection GEMM.
#pragma unroll
  for (int j = 0; j < 4; ++j)
#pragma unroll
    for (int v = 0; v < 8; ++v) {
      int mrow = qrow0 + v + ((lane >> 4) << 3);
      int ncol = h * HD + j * 16 + (lane & 15);
      Ob[(size_t)(b * QLEN + mrow) * DIM + ncol] = (__bf16)(o[j][v] / rsum[v]);
    }
}

// ---------------------------------------------------------------------------
extern "C" void kernel_launch(void* const* d_in, const int* in_sizes, int n_in,
                              void* d_out, int out_size, void* d_ws, size_t ws_size,
                              hipStream_t stream) {
  const float* s_x  = (const float*)d_in[0];
  const float* t_x  = (const float*)d_in[1];
  const float* csp  = (const float*)d_in[2];
  const float* ctp  = (const float*)d_in[3];
  const float* vsp  = (const float*)d_in[4];
  const float* vtp  = (const float*)d_in[5];
  const float* Wq   = (const float*)d_in[6];
  const float* qb   = (const float*)d_in[7];
  const float* Wkv  = (const float*)d_in[8];
  const float* kvb  = (const float*)d_in[9];
  const float* Wp   = (const float*)d_in[10];
  const float* pb   = (const float*)d_in[11];
  float* out = (float*)d_out;

  char* ws = (char*)d_ws;
  size_t off = 0;
  auto alloc = [&](size_t bytes) -> void* {
    void* p = ws + off;
    off = (off + bytes + 255) & ~(size_t)255;
    return p;
  };
  __bf16* Qin  = (__bf16*)alloc((size_t)MQ * DIM * 2);
  __bf16* Sbuf = (__bf16*)alloc((size_t)MS * DIM * 2);
  __bf16* Wqb  = (__bf16*)alloc((size_t)DIM * DIM * 2);
  __bf16* Wkvb = (__bf16*)alloc((size_t)2 * DIM * DIM * 2);
  __bf16* Wpb  = (__bf16*)alloc((size_t)DIM * DIM * 2);
  __bf16* Qbuf = (__bf16*)alloc((size_t)MQ * DIM * 2);
  __bf16* KVb  = (__bf16*)alloc((size_t)MS * 2 * DIM * 2);
  __bf16* Obuf = (__bf16*)alloc((size_t)MQ * DIM * 2);

  // Weight conversion f32 -> bf16 (layout kept [N, K] row-major).
  cvt_bf16_kernel<<<(DIM * DIM / 4 + 255) / 256, 256, 0, stream>>>(Wq, Wqb, DIM * DIM / 4);
  cvt_bf16_kernel<<<(2 * DIM * DIM / 4 + 255) / 256, 256, 0, stream>>>(Wkv, Wkvb, 2 * DIM * DIM / 4);
  cvt_bf16_kernel<<<(DIM * DIM / 4 + 255) / 256, 256, 0, stream>>>(Wp, Wpb, DIM * DIM / 4);

  // Pos-embedded activations.
  prep_qin_kernel<<<MQ * (DIM / 4) / 256, 256, 0, stream>>>(t_x, vsp, vtp, Qin);
  prep_s_kernel<<<MS * (DIM / 4) / 256, 256, 0, stream>>>(s_x, csp, ctp, Sbuf);

  // Q = (Qin @ Wq^T + qb) * SCALE ; KV = Sbuf @ Wkv^T + kvb
  gemm_bf16_kernel<0><<<dim3(MQ / 64, DIM / 128), 128, 0, stream>>>(Qin, Wqb, qb, Qbuf, MQ, DIM);
  gemm_bf16_kernel<1><<<dim3(MS / 64, 2 * DIM / 128), 128, 0, stream>>>(Sbuf, Wkvb, kvb, KVb, MS, 2 * DIM);

  // Flash attention: 96 (b,h) pairs x 49 blocks (2 q-tiles of 16 rows each).
  attention_kernel<<<dim3(B_ * NHEAD, QLEN / 32), 64, 0, stream>>>(Qbuf, KVb, Obuf);

  // Final projection + scatter into (1+VP, B*T, DIM) output.
  gemm_bf16_kernel<2><<<dim3(MQ / 64, DIM / 128), 128, 0, stream>>>(Obuf, Wpb, pb, out, MQ, DIM);

  // t_cls passthrough row.
  cls_copy_kernel<<<(B_ * T_ * DIM / 4 + 255) / 256, 256, 0, stream>>>(t_x, out);
}
